// MultiHeadAttention_23038204575913
// MI455X (gfx1250) — matrix operand order repaired and checked by reading, not compiled, not run on hardware
//
#include <hip/hip_runtime.h>
#include <hip/hip_bf16.h>
#include <stdint.h>

// Problem constants (fixed by the reference)
#define H_B  4
#define H_T  2048
#define H_C  1024
#define H_H  16
#define H_HD 64
#define H_O  1024

// CDNA5 async global->LDS copies (ASYNCcnt path), if the toolchain exposes them
#if defined(__has_builtin)
#  if __has_builtin(__builtin_amdgcn_global_load_async_to_lds_b128)
#    define GEMM_ASYNC 1
#  endif
#endif
#ifndef GEMM_ASYNC
#  define GEMM_ASYNC 0
#endif

typedef __attribute__((ext_vector_type(16))) __bf16 v16bf;
typedef __attribute__((ext_vector_type(8)))  __bf16 v8bf;
typedef __attribute__((ext_vector_type(8)))  float  v8f;

// Exact parameter spelling from the toolchain's async-LDS builtin diagnostic:
// generic pointer to GCC-style int4 vector.
typedef int v4i __attribute__((vector_size(16)));
typedef __attribute__((address_space(3))) v4i as3_v4i;

union BF16x16 { v16bf v; v8bf h[2]; };

__device__ __forceinline__ __bf16 to_bf16(float f) {
  unsigned u = __builtin_bit_cast(unsigned, f);
  u += 0x7fffu + ((u >> 16) & 1u);   // round-to-nearest-even
  unsigned short s = (unsigned short)(u >> 16);
  return __builtin_bit_cast(__bf16, s);
}

__device__ __forceinline__ v8f wmma_bf16(v16bf a, v16bf b, v8f c) {
  // D = A(16x32 bf16) * B(32x16 bf16) + C(16x16 f32)
  return __builtin_amdgcn_wmma_f32_16x16x32_bf16(
      /*neg_a=*/false, a, /*neg_b=*/false, b,
      /*c_mod=*/(short)0, c, /*reuse_a=*/false, /*reuse_b=*/false);
}

// ---------------------------------------------------------------------------
// Pass 0a: f32 -> bf16 elementwise
// ---------------------------------------------------------------------------
__global__ void cvt_bf16_kernel(const float* __restrict__ in,
                                __bf16* __restrict__ out, int n) {
  int i = blockIdx.x * blockDim.x + threadIdx.x;
  if (i < n) out[i] = to_bf16(in[i]);
}

// ---------------------------------------------------------------------------
// Pass 0b: W (1024x1024 f32, row-major N x K) -> Wt (K x N bf16)
// ---------------------------------------------------------------------------
__global__ void transpose_bf16_kernel(const float* __restrict__ W,
                                      __bf16* __restrict__ Wt) {
  int tid = blockIdx.x * blockDim.x + threadIdx.x;   // 0 .. 1M-1
  int k = tid >> 10, n = tid & 1023;
  Wt[tid] = to_bf16(W[(size_t)n * 1024 + k]);
}

// ---------------------------------------------------------------------------
// bf16 WMMA GEMM:  C[M,N] = scale * A[MxK] @ Wt[KxN]   (Wt is K-major)
//   256 threads = 8 waves; block tile 128x64; K-tile 64 (double-buffered LDS).
//   Wave (mw = w&3, nw = w>>2) computes 32x32 = 2x2 WMMA tiles.
//   mode 0: f32 row-major output                      [final projection]
//   mode 1: bf16 scatter to (B,H,T,HD)                [q, v]
//   mode 2: bf16 scatter to (B,H,HD,T)                [k transposed]
// ---------------------------------------------------------------------------
__global__ __launch_bounds__(256) void gemm_bf16_wmma(
    const __bf16* __restrict__ A, const __bf16* __restrict__ Wt,
    void* __restrict__ out, int M, int N, int K, float scale, int mode) {
  __shared__ __bf16 As[2][128 * 64];   // 16 KB per buffer
  __shared__ __bf16 Bs[2][64 * 64];    //  8 KB per buffer

  const int tid  = threadIdx.x;
  const int wave = tid >> 5, lane = tid & 31;
  const int half = lane >> 4, lr = lane & 15;
  const int mw = wave & 3, nw = wave >> 2;
  const int mbase = blockIdx.x * 128, nbase = blockIdx.y * 64;

  v8f acc[2][2] = {};
  const int KT = K >> 6;

  // Per-thread copy chunks (16B = 8 bf16). A tile: 1024 chunks, B tile: 512.
#if GEMM_ASYNC
  auto issue_tile = [&](int k0, int buf) {
#pragma unroll
    for (int i = 0; i < 4; ++i) {
      int c = tid + i * 256, r = c >> 3, co = (c & 7) * 8;
      __builtin_amdgcn_global_load_async_to_lds_b128(
          (v4i*)(A + (size_t)(mbase + r) * K + k0 + co),
          (as3_v4i*)&As[buf][r * 64 + co], 0, 0);
    }
#pragma unroll
    for (int i = 0; i < 2; ++i) {
      int c = tid + i * 256, r = c >> 3, co = (c & 7) * 8;
      __builtin_amdgcn_global_load_async_to_lds_b128(
          (v4i*)(Wt + (size_t)(k0 + r) * N + nbase + co),
          (as3_v4i*)&Bs[buf][r * 64 + co], 0, 0);
    }
  };
  issue_tile(0, 0);
  asm volatile("s_wait_asynccnt 0x0" ::: "memory");
  __syncthreads();
#else
  v8bf ra[4], rb[2];
  auto load_tile = [&](int k0) {
#pragma unroll
    for (int i = 0; i < 4; ++i) {
      int c = tid + i * 256, r = c >> 3, co = (c & 7) * 8;
      ra[i] = *(const v8bf*)(A + (size_t)(mbase + r) * K + k0 + co);
    }
#pragma unroll
    for (int i = 0; i < 2; ++i) {
      int c = tid + i * 256, r = c >> 3, co = (c & 7) * 8;
      rb[i] = *(const v8bf*)(Wt + (size_t)(k0 + r) * N + nbase + co);
    }
  };
  auto store_tile = [&](int buf) {
#pragma unroll
    for (int i = 0; i < 4; ++i) {
      int c = tid + i * 256, r = c >> 3, co = (c & 7) * 8;
      *(v8bf*)&As[buf][r * 64 + co] = ra[i];
    }
#pragma unroll
    for (int i = 0; i < 2; ++i) {
      int c = tid + i * 256, r = c >> 3, co = (c & 7) * 8;
      *(v8bf*)&Bs[buf][r * 64 + co] = rb[i];
    }
  };
  load_tile(0);
  store_tile(0);
  __syncthreads();
#endif

  for (int kt = 0; kt < KT; ++kt) {
    const int cur = kt & 1;
#if GEMM_ASYNC
    if (kt + 1 < KT) issue_tile((kt + 1) << 6, cur ^ 1);  // overlap with WMMA
#else
    if (kt + 1 < KT) load_tile((kt + 1) << 6);            // loads in flight
#endif

#pragma unroll
    for (int ks = 0; ks < 64; ks += 32) {
      BF16x16 afr[2], bfr[2];
#pragma unroll
      for (int ms = 0; ms < 2; ++ms) {
        // A frag (16-bit 16x32 layout): row, chunks at ks+half*8 and +16
        const __bf16* p = &As[cur][(mw * 32 + ms * 16 + lr) * 64 + ks + half * 8];
        afr[ms].h[0] = *(const v8bf*)p;
        afr[ms].h[1] = *(const v8bf*)(p + 16);
      }
#pragma unroll
      for (int nt = 0; nt < 2; ++nt) {
        // B frag: lane = N column, K rows half*8+i and 16+half*8+i
        const __bf16* p = &Bs[cur][(ks + half * 8) * 64 + nw * 32 + nt * 16 + lr];
        for (int i = 0; i < 8; ++i) {
          bfr[nt].v[i] = p[i * 64];
          bfr[nt].v[8 + i] = p[(16 + i) * 64];
        }
      }
#pragma unroll
      for (int ms = 0; ms < 2; ++ms)
#pragma unroll
        for (int nt = 0; nt < 2; ++nt)
          acc[ms][nt] = wmma_bf16(afr[ms].v, bfr[nt].v, acc[ms][nt]);
    }

#if GEMM_ASYNC
    asm volatile("s_wait_asynccnt 0x0" ::: "memory");
#else
    if (kt + 1 < KT) store_tile(cur ^ 1);
#endif
    __syncthreads();
  }

  // C/D layout: VGPR v, lane L -> M = v + 8*(L/16), N = L%16
#pragma unroll
  for (int ms = 0; ms < 2; ++ms) {
#pragma unroll
    for (int nt = 0; nt < 2; ++nt) {
#pragma unroll
      for (int vv = 0; vv < 8; ++vv) {
        int row = mbase + mw * 32 + ms * 16 + vv + half * 8;
        int col = nbase + nw * 32 + nt * 16 + lr;
        float val = acc[ms][nt][vv] * scale;
        if (mode == 0) {
          ((float*)out)[(size_t)row * N + col] = val;
        } else {
          int b = row >> 11, t = row & 2047;   // M = B*T, T=2048
          int h = col >> 6,  d = col & 63;     // N = H*HD, HD=64
          __bf16 bv = to_bf16(val);
          if (mode == 1)
            ((__bf16*)out)[(((size_t)b * H_H + h) * H_T + t) * H_HD + d] = bv;
          else
            ((__bf16*)out)[(((size_t)b * H_H + h) * H_HD + d) * H_T + t] = bv;
        }
      }
    }
  }
}

// ---------------------------------------------------------------------------
// Flash attention:  per block = one (b, h, 64-row q tile).
//   q: (B,H,T,HD)  kT: (B,H,HD,T)  v: (B,H,T,HD)  all bf16 scaled by C^-1/2
//   att out: (B,T,H*HD) bf16
// ---------------------------------------------------------------------------
__global__ __launch_bounds__(256) void attn_flash_wmma(
    const __bf16* __restrict__ qp, const __bf16* __restrict__ kT,
    const __bf16* __restrict__ vp, __bf16* __restrict__ att) {
  __shared__ __bf16 qs[64 * 64];
  __shared__ float  Ss[64 * 65];     // padded: conflict-free row reductions
  __shared__ __bf16 Ps[64 * 64];
  __shared__ float  mrow[64], lrow[64], arw[64];

  int tid  = threadIdx.x;
  int wave = tid >> 5, lane = tid & 31;
  int half = lane >> 4, lr = lane & 15;
  int qi = wave >> 1, ch = wave & 1;

  int qtile = blockIdx.x, h = blockIdx.y, b = blockIdx.z;
  int qbase = qtile * 64;
  size_t bh = (size_t)b * H_H + h;

  {
    const __bf16* qg = qp + (bh * H_T + qbase) * H_HD;
    int i0 = tid * 16;
    *(v8bf*)&qs[i0]     = *(const v8bf*)&qg[i0];
    *(v8bf*)&qs[i0 + 8] = *(const v8bf*)&qg[i0 + 8];
  }
  if (tid < 64) { mrow[tid] = -3.0e38f; lrow[tid] = 0.f; }
  __syncthreads();

  v8f accO[2] = {};

  for (int j = 0; j <= qtile; ++j) {
    int key0 = j * 64;

    // ---- S = (q @ kT) * hd^-1/2, 64x64 chunk ----
    v8f accS[2] = {};
#pragma unroll
    for (int ks = 0; ks < 64; ks += 32) {
      BF16x16 af;
      const __bf16* ap = &qs[(qi * 16 + lr) * 64 + ks + half * 8];
      af.h[0] = *(const v8bf*)ap;
      af.h[1] = *(const v8bf*)(ap + 16);
      const __bf16* kbase = kT + (bh * H_HD + ks + half * 8) * H_T + key0 + ch * 32 + lr;
#pragma unroll
      for (int nt = 0; nt < 2; ++nt) {
        BF16x16 bfr;
        const __bf16* bp = kbase + nt * 16;
        for (int i = 0; i < 8; ++i) {
          bfr.v[i] = bp[(size_t)i * H_T];
          bfr.v[8 + i] = bp[(size_t)(16 + i) * H_T];
        }
        accS[nt] = wmma_bf16(af.v, bfr.v, accS[nt]);
      }
    }
#pragma unroll
    for (int nt = 0; nt < 2; ++nt) {
#pragma unroll
      for (int vv = 0; vv < 8; ++vv) {
        int mloc = qi * 16 + vv + half * 8;
        int nloc = ch * 32 + nt * 16 + lr;
        float s = accS[nt][vv] * 0.125f;          // hd^-1/2 = 1/8
        if (key0 + nloc > qbase + mloc) s = -3.0e38f;
        Ss[mloc * 65 + nloc] = s;
      }
    }
    __syncthreads();

    // ---- online softmax: one thread per q row ----
    if (tid < 64) {
      float mo = mrow[tid], mx = mo;
      for (int c = 0; c < 64; ++c) mx = fmaxf(mx, Ss[tid * 65 + c]);
      float al = __expf(mo - mx);
      float sum = 0.f;
      for (int c = 0; c < 64; ++c) {
        float p = __expf(Ss[tid * 65 + c] - mx);
        sum += p;
        Ps[tid * 64 + c] = to_bf16(p);
      }
      mrow[tid] = mx;
      lrow[tid] = lrow[tid] * al + sum;
      arw[tid]  = al;
    }
    __syncthreads();

    // ---- rescale O, then O += P @ V_chunk ----
#pragma unroll
    for (int nt = 0; nt < 2; ++nt)
#pragma unroll
      for (int vv = 0; vv < 8; ++vv)
        accO[nt][vv] *= arw[qi * 16 + vv + half * 8];

#pragma unroll
    for (int ks = 0; ks < 64; ks += 32) {
      BF16x16 af;
      const __bf16* pp = &Ps[(qi * 16 + lr) * 64 + ks + half * 8];
      af.h[0] = *(const v8bf*)pp;
      af.h[1] = *(const v8bf*)(pp + 16);
      const __bf16* vbase = vp + (bh * H_T + key0 + ks + half * 8) * H_HD + ch * 32 + lr;
#pragma unroll
      for (int nt = 0; nt < 2; ++nt) {
        BF16x16 bfr;
        const __bf16* bp = vbase + nt * 16;
        for (int i = 0; i < 8; ++i) {
          bfr.v[i] = bp[i * H_HD];
          bfr.v[8 + i] = bp[(16 + i) * H_HD];
        }
        accO[nt] = wmma_bf16(af.v, bfr.v, accO[nt]);
      }
    }
    __syncthreads();   // protect Ss/Ps before next chunk overwrites them
  }

  // ---- epilogue: O / l -> att (B,T,H*HD) ----
#pragma unroll
  for (int nt = 0; nt < 2; ++nt) {
#pragma unroll
    for (int vv = 0; vv < 8; ++vv) {
      int mloc = qi * 16 + vv + half * 8;
      int dcol = ch * 32 + nt * 16 + lr;
      float o = accO[nt][vv] / lrow[mloc];
      att[((size_t)b * H_T + qbase + mloc) * (H_H * H_HD) + h * H_HD + dcol] =
          to_bf16(o);
    }
  }
}

// ---------------------------------------------------------------------------
// Host launcher
// ---------------------------------------------------------------------------
extern "C" void kernel_launch(void* const* d_in, const int* in_sizes, int n_in,
                              void* d_out, int out_size, void* d_ws,
                              size_t ws_size, hipStream_t stream) {
  (void)in_sizes; (void)n_in; (void)out_size; (void)ws_size;
  const float* x  = (const float*)d_in[0];
  const float* Wk = (const float*)d_in[1];
  const float* Wq = (const float*)d_in[2];
  const float* Wv = (const float*)d_in[3];
  const float* Wp = (const float*)d_in[4];
  float* out = (float*)d_out;

  char* ws = (char*)d_ws;
  size_t off = 0;
  auto carve = [&](size_t bytes) -> char* {
    char* p = ws + off;
    off += (bytes + 255) & ~(size_t)255;
    return p;
  };

  const size_t MT = (size_t)H_B * H_T;  // 8192 rows
  __bf16* xb  = (__bf16*)carve(MT * H_C * 2);
  __bf16* WtK = (__bf16*)carve((size_t)H_C * H_O * 2);
  __bf16* WtQ = (__bf16*)carve((size_t)H_C * H_O * 2);
  __bf16* WtV = (__bf16*)carve((size_t)H_C * H_O * 2);
  __bf16* WtP = (__bf16*)carve((size_t)H_O * H_O * 2);
  __bf16* qb  = (__bf16*)carve(MT * H_O * 2);            // q  (B,H,T,HD)
  __bf16* kTb = (__bf16*)carve(MT * H_O * 2);            // kT (B,H,HD,T)
  __bf16* vb  = (__bf16*)carve(MT * H_O * 2);            // v  (B,H,T,HD)
  __bf16* ab  = (__bf16*)carve(MT * H_O * 2);            // attn out (B,T,O)

  int nx = (int)(MT * H_C);
  cvt_bf16_kernel<<<nx / 256, 256, 0, stream>>>(x, xb, nx);
  int nw = H_C * H_O;
  transpose_bf16_kernel<<<nw / 256, 256, 0, stream>>>(Wk, WtK);
  transpose_bf16_kernel<<<nw / 256, 256, 0, stream>>>(Wq, WtQ);
  transpose_bf16_kernel<<<nw / 256, 256, 0, stream>>>(Wv, WtV);
  transpose_bf16_kernel<<<nw / 256, 256, 0, stream>>>(Wp, WtP);

  dim3 gg((unsigned)(MT / 128), H_O / 64);
  gemm_bf16_wmma<<<gg, 256, 0, stream>>>(xb, WtQ, qb,  (int)MT, H_O, H_C,
                                         0.03125f, 1);
  gemm_bf16_wmma<<<gg, 256, 0, stream>>>(xb, WtK, kTb, (int)MT, H_O, H_C,
                                         0.03125f, 2);
  gemm_bf16_wmma<<<gg, 256, 0, stream>>>(xb, WtV, vb,  (int)MT, H_O, H_C,
                                         0.03125f, 1);

  attn_flash_wmma<<<dim3(H_T / 64, H_H, H_B), 256, 0, stream>>>(qb, kTb, vb,
                                                                ab);

  gemm_bf16_wmma<<<gg, 256, 0, stream>>>(ab, WtP, out, (int)MT, H_O, H_O,
                                         0.03125f, 0);
}
